// Multi_Head_Attention_45217415693126
// MI455X (gfx1250) — matrix operand order repaired and checked by reading, not compiled, not run on hardware
//
#include <hip/hip_runtime.h>

// ---------------------------------------------------------------------------
// Multi-head attention forward on gfx1250 (wave32, WMMA bf16 16x16x32).
// B=4, N=2048, D=1024, H=16, DH=64.
// ---------------------------------------------------------------------------

typedef __attribute__((ext_vector_type(16))) __bf16 v16bf;
typedef __attribute__((ext_vector_type(8)))  __bf16 v8bf;
typedef __attribute__((ext_vector_type(8)))  float  v8f;

#define BB   4
#define NN   2048
#define DD   1024
#define HH   16
#define DHH  64
#define MROWS (BB * NN)          // 8192
#define N3   (3 * DD)            // 3072
#define SCALE 0.125f             // 64^-0.5

__device__ __forceinline__ __bf16 f2bf(float f) {
    union { float f; unsigned u; } v; v.f = f;
    unsigned r = v.u + 0x7FFFu + ((v.u >> 16) & 1u);   // round-to-nearest-even
    unsigned short h = (unsigned short)(r >> 16);
    return __builtin_bit_cast(__bf16, h);
}

// ---------------------------------------------------------------------------
// Stage 0: fp32 -> bf16 converts (copy + transpose for weights)
// ---------------------------------------------------------------------------
__global__ void __launch_bounds__(256) cvt_copy(const float* __restrict__ src,
                                                __bf16* __restrict__ dst, int n) {
    int i = blockIdx.x * 256 + threadIdx.x;
    if (i < n) dst[i] = f2bf(src[i]);
}

// dst[n*K + k] = src[k*Ncols + n]   (store weights column-major in bf16)
__global__ void __launch_bounds__(256) cvt_transpose(const float* __restrict__ src,
                                                     __bf16* __restrict__ dst,
                                                     int K, int Ncols) {
    int i = blockIdx.x * 256 + threadIdx.x;
    if (i < K * Ncols) {
        int n = i / K;
        int k = i - n * K;
        dst[i] = f2bf(src[(size_t)k * Ncols + n]);
    }
}

// ---------------------------------------------------------------------------
// Stage 1: QKV projection. C[8192,3072] = Xb @ WqkvT + b, scattered to
// q_bf[b,h,n,dh] (scaled), k_bf[b,h,n,dh], vT_bf[b,h,dh,n].
// One 16x64 strip per wave: A fragment reused across 4 B fragments
// (4 WMMAs per K-step of 32).
// ---------------------------------------------------------------------------
__global__ void __launch_bounds__(256) qkv_gemm(const __bf16* __restrict__ xb,
                                                const __bf16* __restrict__ wT,
                                                const float*  __restrict__ bias,
                                                __bf16* __restrict__ qb,
                                                __bf16* __restrict__ kb,
                                                __bf16* __restrict__ vT) {
    const int w    = threadIdx.x >> 5;
    const int lane = threadIdx.x & 31;
    const int lm   = lane & 15;
    const int lh   = lane >> 4;
    const int tile = blockIdx.x * 8 + w;      // 16x64 strips
    const int mt   = tile / (N3 / 64);        // / 48
    const int nt   = tile - mt * (N3 / 64);
    const int m0   = mt * 16;
    const int n0   = nt * 64;

    const __bf16* arow = xb + (size_t)(m0 + lm) * DD;
    const __bf16* bcol0 = wT + (size_t)(n0 + lm) * DD;

    v8f acc[4];
#pragma unroll
    for (int st = 0; st < 4; ++st) acc[st] = (v8f){};

#pragma unroll 2
    for (int k0 = 0; k0 < DD; k0 += 32) {
        v8bf a0 = *(const v8bf*)(arow + k0 + 8 * lh);
        v8bf a1 = *(const v8bf*)(arow + k0 + 16 + 8 * lh);
        v16bf a;
#pragma unroll
        for (int e = 0; e < 8; ++e) { a[e] = a0[e]; a[e + 8] = a1[e]; }
#pragma unroll
        for (int st = 0; st < 4; ++st) {
            const __bf16* bcol = bcol0 + (size_t)(st * 16) * DD;
            v8bf b0 = *(const v8bf*)(bcol + k0 + 8 * lh);
            v8bf b1 = *(const v8bf*)(bcol + k0 + 16 + 8 * lh);
            v16bf b;
#pragma unroll
            for (int e = 0; e < 8; ++e) { b[e] = b0[e]; b[e + 8] = b1[e]; }
            acc[st] = __builtin_amdgcn_wmma_f32_16x16x32_bf16(
                false, a, false, b, (short)0, acc[st], false, false);
        }
    }

#pragma unroll
    for (int st = 0; st < 4; ++st) {
        const int col   = n0 + st * 16 + lm;   // 0..3071
        const float bv  = bias[col];
        const int three = col / DD;            // 0=q 1=k 2=v (uniform per wave)
        const int h     = (col - three * DD) / DHH;
        const int dh    = col & (DHH - 1);
#pragma unroll
        for (int r = 0; r < 8; ++r) {
            const int row = m0 + r + 8 * lh;         // 0..8191
            const int bi  = row / NN;
            const int nn  = row - bi * NN;
            const size_t bh = (size_t)bi * HH + h;
            const float val = acc[st][r] + bv;
            if (three == 0)      qb[(bh * NN + nn) * DHH + dh] = f2bf(val * SCALE);
            else if (three == 1) kb[(bh * NN + nn) * DHH + dh] = f2bf(val);
            else                 vT[(bh * DHH + dh) * NN + nn] = f2bf(val);
        }
    }
}

// ---------------------------------------------------------------------------
// Stage 2: flash-style attention. One wave per 16-query block of one (b,h).
// Streams keys in chunks of 32: 4 WMMAs for S, online softmax, P via LDS,
// 4 WMMAs for P@V (DH=64 -> 4 accumulator tiles).
// ---------------------------------------------------------------------------
__global__ void __launch_bounds__(256) attn_flash(const __bf16* __restrict__ qb,
                                                  const __bf16* __restrict__ kb,
                                                  const __bf16* __restrict__ vT,
                                                  __bf16* __restrict__ ob) {
    __shared__ __bf16 Plds[8][16][40];   // stride 40 bf16: 16B-aligned rows, conflict-free
    const int w    = threadIdx.x >> 5;
    const int lane = threadIdx.x & 31;
    const int lm   = lane & 15;
    const int lh   = lane >> 4;
    const int tile = blockIdx.x * 8 + w;
    const int bh   = tile >> 7;              // / (N/16 = 128)
    const int q0   = (tile & 127) << 4;

    // Q block as two A-operands (dh 0..31, 32..63), loaded once.
    const __bf16* Qrow = qb + ((size_t)bh * NN + q0 + lm) * DHH;
    v16bf A0, A1;
    {
        v8bf t0 = *(const v8bf*)(Qrow + 8 * lh);
        v8bf t1 = *(const v8bf*)(Qrow + 16 + 8 * lh);
        v8bf t2 = *(const v8bf*)(Qrow + 32 + 8 * lh);
        v8bf t3 = *(const v8bf*)(Qrow + 48 + 8 * lh);
#pragma unroll
        for (int e = 0; e < 8; ++e) {
            A0[e] = t0[e]; A0[e + 8] = t1[e];
            A1[e] = t2[e]; A1[e + 8] = t3[e];
        }
    }

    v8f accO[4];
    float mrow[8], lrow[8], corr[8];
#pragma unroll
    for (int dt = 0; dt < 4; ++dt) accO[dt] = (v8f){};
#pragma unroll
    for (int r = 0; r < 8; ++r) { mrow[r] = -3.0e38f; lrow[r] = 0.0f; }

    const __bf16* Kbh = kb + (size_t)bh * NN * DHH;
    const __bf16* Vbh = vT + (size_t)bh * DHH * NN;

    for (int j = 0; j < NN; j += 32) {
        // ---- scores: two 16-key tiles, K-dim = dh split 0..31 / 32..63 ----
        v8f s[2];
#pragma unroll
        for (int t = 0; t < 2; ++t) {
            const __bf16* krow = Kbh + (size_t)(j + t * 16 + lm) * DHH;
            v8bf u0 = *(const v8bf*)(krow + 8 * lh);
            v8bf u1 = *(const v8bf*)(krow + 16 + 8 * lh);
            v8bf u2 = *(const v8bf*)(krow + 32 + 8 * lh);
            v8bf u3 = *(const v8bf*)(krow + 48 + 8 * lh);
            v16bf B0, B1;
#pragma unroll
            for (int e = 0; e < 8; ++e) {
                B0[e] = u0[e]; B0[e + 8] = u1[e];
                B1[e] = u2[e]; B1[e + 8] = u3[e];
            }
            v8f z = {};
            z = __builtin_amdgcn_wmma_f32_16x16x32_bf16(false, A0, false, B0,
                                                        (short)0, z, false, false);
            z = __builtin_amdgcn_wmma_f32_16x16x32_bf16(false, A1, false, B1,
                                                        (short)0, z, false, false);
            s[t] = z;
        }

        // ---- online softmax (row reductions across 16-lane halves) ----
#pragma unroll
        for (int r = 0; r < 8; ++r) {
            float mx = fmaxf(s[0][r], s[1][r]);
            mx = fmaxf(mx, __shfl_xor(mx, 1, 32));
            mx = fmaxf(mx, __shfl_xor(mx, 2, 32));
            mx = fmaxf(mx, __shfl_xor(mx, 4, 32));
            mx = fmaxf(mx, __shfl_xor(mx, 8, 32));
            const float mnew = fmaxf(mrow[r], mx);
            corr[r] = __expf(mrow[r] - mnew);
            mrow[r] = mnew;
            const float p0 = __expf(s[0][r] - mnew);
            const float p1 = __expf(s[1][r] - mnew);
            s[0][r] = p0; s[1][r] = p1;
            float rs = p0 + p1;
            rs += __shfl_xor(rs, 1, 32);
            rs += __shfl_xor(rs, 2, 32);
            rs += __shfl_xor(rs, 4, 32);
            rs += __shfl_xor(rs, 8, 32);
            lrow[r] = lrow[r] * corr[r] + rs;
        }

        // ---- C-layout -> A-layout via LDS (same-wave, s_wait_dscnt) ----
#pragma unroll
        for (int r = 0; r < 8; ++r) {
            const int row = r + 8 * lh;
            Plds[w][row][lm]      = f2bf(s[0][r]);
            Plds[w][row][16 + lm] = f2bf(s[1][r]);
        }
        asm volatile("s_wait_dscnt 0" ::: "memory");
        v16bf Pa;
        {
            const __bf16* prow = &Plds[w][lm][0];
            v8bf p0 = *(const v8bf*)(prow + 8 * lh);
            v8bf p1 = *(const v8bf*)(prow + 16 + 8 * lh);
#pragma unroll
            for (int e = 0; e < 8; ++e) { Pa[e] = p0[e]; Pa[e + 8] = p1[e]; }
        }

        // ---- O += P @ V over 4 dh tiles (V stored [dh][key]: contiguous) ----
#pragma unroll
        for (int dt = 0; dt < 4; ++dt) {
            v8f c = accO[dt];
#pragma unroll
            for (int r = 0; r < 8; ++r) c[r] *= corr[r];
            const __bf16* vrow = Vbh + (size_t)(dt * 16 + lm) * NN + j + 8 * lh;
            v8bf v0 = *(const v8bf*)(vrow);
            v8bf v1 = *(const v8bf*)(vrow + 16);
            v16bf Vb;
#pragma unroll
            for (int e = 0; e < 8; ++e) { Vb[e] = v0[e]; Vb[e + 8] = v1[e]; }
            accO[dt] = __builtin_amdgcn_wmma_f32_16x16x32_bf16(false, Pa, false, Vb,
                                                               (short)0, c, false, false);
        }
    }

    // ---- epilogue: normalize and write bf16 O in [b,n, h*64+dh] layout ----
    const int bi = bh >> 4, h = bh & 15;
#pragma unroll
    for (int dt = 0; dt < 4; ++dt) {
#pragma unroll
        for (int r = 0; r < 8; ++r) {
            const int qrow = q0 + r + 8 * lh;
            const float val = accO[dt][r] / lrow[r];
            ob[((size_t)(bi * NN + qrow)) * DD + h * DHH + dt * 16 + lm] = f2bf(val);
        }
    }
}

// ---------------------------------------------------------------------------
// Stage 3: output projection. d_out[8192,1024] = Ob @ WoutT + b_out (fp32).
// One 16x64 strip per wave (A reuse x4).
// ---------------------------------------------------------------------------
__global__ void __launch_bounds__(256) out_gemm(const __bf16* __restrict__ ob,
                                                const __bf16* __restrict__ wT,
                                                const float*  __restrict__ bias,
                                                float* __restrict__ out) {
    const int w    = threadIdx.x >> 5;
    const int lane = threadIdx.x & 31;
    const int lm   = lane & 15;
    const int lh   = lane >> 4;
    const int tile = blockIdx.x * 8 + w;     // 16x64 strips
    const int mt   = tile >> 4;              // / (D/64 = 16)
    const int nt   = tile & 15;
    const int m0   = mt * 16;
    const int n0   = nt * 64;

    const __bf16* arow  = ob + (size_t)(m0 + lm) * DD;
    const __bf16* bcol0 = wT + (size_t)(n0 + lm) * DD;

    v8f acc[4];
#pragma unroll
    for (int st = 0; st < 4; ++st) acc[st] = (v8f){};

#pragma unroll 2
    for (int k0 = 0; k0 < DD; k0 += 32) {
        v8bf a0 = *(const v8bf*)(arow + k0 + 8 * lh);
        v8bf a1 = *(const v8bf*)(arow + k0 + 16 + 8 * lh);
        v16bf a;
#pragma unroll
        for (int e = 0; e < 8; ++e) { a[e] = a0[e]; a[e + 8] = a1[e]; }
#pragma unroll
        for (int st = 0; st < 4; ++st) {
            const __bf16* bcol = bcol0 + (size_t)(st * 16) * DD;
            v8bf b0 = *(const v8bf*)(bcol + k0 + 8 * lh);
            v8bf b1 = *(const v8bf*)(bcol + k0 + 16 + 8 * lh);
            v16bf b;
#pragma unroll
            for (int e = 0; e < 8; ++e) { b[e] = b0[e]; b[e + 8] = b1[e]; }
            acc[st] = __builtin_amdgcn_wmma_f32_16x16x32_bf16(
                false, a, false, b, (short)0, acc[st], false, false);
        }
    }

#pragma unroll
    for (int st = 0; st < 4; ++st) {
        const float bv = bias[n0 + st * 16 + lm];
#pragma unroll
        for (int r = 0; r < 8; ++r)
            out[(size_t)(m0 + r + 8 * lh) * DD + n0 + st * 16 + lm] = acc[st][r] + bv;
    }
}

// ---------------------------------------------------------------------------
// Host launcher
// ---------------------------------------------------------------------------
extern "C" void kernel_launch(void* const* d_in, const int* in_sizes, int n_in,
                              void* d_out, int out_size, void* d_ws, size_t ws_size,
                              hipStream_t stream) {
    (void)in_sizes; (void)n_in; (void)out_size; (void)ws_size;
    const float* x    = (const float*)d_in[0];   // [B,N,D]
    const float* Wqkv = (const float*)d_in[1];   // [D,3D]
    const float* bqkv = (const float*)d_in[2];   // [3D]
    const float* Wout = (const float*)d_in[3];   // [D,D]
    const float* bout = (const float*)d_in[4];   // [D]
    float* out = (float*)d_out;                  // [B,N,D]

    // workspace carve-up (bf16 elements)
    __bf16* xb    = (__bf16*)d_ws;                 // 8,388,608  x in bf16
    __bf16* wqkvT = xb    + (size_t)MROWS * DD;    // 3,145,728  Wqkv^T
    __bf16* woutT = wqkvT + (size_t)DD * N3;       // 1,048,576  Wout^T
    __bf16* qb    = woutT + (size_t)DD * DD;       // 8,388,608  Q scaled
    __bf16* kbuf  = qb    + (size_t)MROWS * DD;    // 8,388,608  K
    __bf16* vT    = kbuf  + (size_t)MROWS * DD;    // 8,388,608  V^T
    __bf16* ob    = vT    + (size_t)MROWS * DD;    // 8,388,608  attn output

    const int nX = MROWS * DD;            // 8,388,608
    cvt_copy<<<nX / 256, 256, 0, stream>>>(x, xb, nX);
    cvt_transpose<<<(DD * N3) / 256, 256, 0, stream>>>(Wqkv, wqkvT, DD, N3);
    cvt_transpose<<<(DD * DD) / 256, 256, 0, stream>>>(Wout, woutT, DD, DD);

    // 512 * 48 strips / 8 waves per block
    qkv_gemm<<<(MROWS / 16) * (N3 / 64) / 8, 256, 0, stream>>>(
        xb, wqkvT, bqkv, qb, kbuf, vT);

    // B*H*(N/16) = 8192 query blocks / 8 waves per block
    attn_flash<<<BB * HH * (NN / 16) / 8, 256, 0, stream>>>(qb, kbuf, vT, ob);

    // 512 * 16 strips / 8 waves per block
    out_gemm<<<(MROWS / 16) * (DD / 64) / 8, 256, 0, stream>>>(
        ob, woutT, bout, out);
}
